// VectorQuantizer_85478439125909
// MI455X (gfx1250) — matrix operand order repaired and checked by reading, not compiled, not run on hardware
//
#include <hip/hip_runtime.h>

#define N_TOKENS 16384
#define DIM 512
#define CBSZ 8192
#define KSTEPS 16          // 512 / 32
#define NCHUNKS 128        // 8192 / 64 total; each wave sweeps half (64)
#define EPS 1e-10f

typedef __attribute__((ext_vector_type(16))) __bf16 v16bf;
typedef __attribute__((ext_vector_type(8)))  float  v8f;

// ---------------- bf16 fragment loaders (pure b128 loads, no cvt) ----------------

// A fragment (16x32): lane half `sel` holds K = {sel*8..+7} and {16+sel*8..+7}
// of row (lane&15). p is pre-offset by sel*8. Two 16-byte loads.
__device__ inline v16bf load_frag_a(const __bf16* __restrict__ p) {
    union { v16bf v; float4 f[2]; } u;
    u.f[0] = *(const float4*)(p);
    u.f[1] = *(const float4*)(p + 16);
    return u.v;
}

// B fragment (32x16): lane half `sel` holds K = {sel*8..+7} and {16+sel*8..+7}
// of column (lane&15) (p pre-offset by sel*8). Two 16-byte loads.
__device__ inline v16bf load_frag_b(const __bf16* __restrict__ p) {
    union { v16bf v; float4 f[2]; } u;
    u.f[0] = *(const float4*)(p);
    u.f[1] = *(const float4*)(p + 16);
    return u.v;
}

// One k-step's operands: a-hi, a-lo and the 4 column tiles' b-hi/b-lo.
struct KStage {
    v16bf ah, al;
    v16bf bh[4], bl[4];
};

__device__ inline void load_stage(KStage& st, int k,
        const __bf16* __restrict__ zhr, const __bf16* __restrict__ zlr,
        const __bf16* __restrict__ cbhi, const __bf16* __restrict__ cblo,
        size_t bbase)
{
    st.ah = load_frag_a(zhr + k * 32);
    st.al = load_frag_a(zlr + k * 32);
#pragma unroll
    for (int t = 0; t < 4; ++t) {
        const size_t boff = bbase + (size_t)t * 16 * DIM + k * 32;
        st.bh[t] = load_frag_b(cbhi + boff);
        st.bl[t] = load_frag_b(cblo + boff);
    }
}

// split-bf16 triple product for one stage: 12 WMMAs, ~f32 accuracy.
__device__ inline void stage_wmma(const KStage& st, v8f (&acc)[4]) {
#pragma unroll
    for (int t = 0; t < 4; ++t) {
        acc[t] = __builtin_amdgcn_wmma_f32_16x16x32_bf16(false, st.ah, false, st.bh[t], (short)0, acc[t], false, false);
        acc[t] = __builtin_amdgcn_wmma_f32_16x16x32_bf16(false, st.al, false, st.bh[t], (short)0, acc[t], false, false);
        acc[t] = __builtin_amdgcn_wmma_f32_16x16x32_bf16(false, st.ah, false, st.bl[t], (short)0, acc[t], false, false);
    }
}

// Software-pipelined 64-column chunk: stage k+1's loads are issued before
// stage k's 12 WMMAs, so the load wait hides behind the XDL pipeline.
__device__ inline void chunk_gemm(
        const __bf16* __restrict__ zhr, const __bf16* __restrict__ zlr,
        const __bf16* __restrict__ cbhi, const __bf16* __restrict__ cblo,
        size_t bbase, v8f (&acc)[4])
{
    KStage s0, s1;
    load_stage(s0, 0, zhr, zlr, cbhi, cblo, bbase);
#pragma unroll
    for (int k = 0; k < KSTEPS; ++k) {
        KStage& cur = (k & 1) ? s1 : s0;
        KStage& nxt = (k & 1) ? s0 : s1;
        if (k + 1 < KSTEPS)
            load_stage(nxt, k + 1, zhr, zlr, cbhi, cblo, bbase);
        stage_wmma(cur, acc);
    }
}

// ---------------- prep: split f32 -> bf16 hi/lo planes ----------------
__global__ __launch_bounds__(256) void vq_split(
        const float* __restrict__ x, __bf16* __restrict__ hi,
        __bf16* __restrict__ lo, int n)
{
    const int i = blockIdx.x * blockDim.x + threadIdx.x;
    if (i < n) {
        const float v = x[i];
        const __bf16 h = (__bf16)v;
        hi[i] = h;
        lo[i] = (__bf16)(v - (float)h);
    }
}

// ---------------- Pass 1: GEMM + fused argmax + online logsumexp ----------------
// Each wave owns 16 rows and HALF the codebook (64 chunks); two waves per row
// block. logits l = 200 * dot (argmin dists == argmax l). Per-(row,half)
// partial (m, s, w, best, bidx) written to ws; merged by vq_merge.
__global__ __launch_bounds__(128) void vq_pass1(
        const __bf16* __restrict__ zhi, const __bf16* __restrict__ zlo,
        const __bf16* __restrict__ cbhi, const __bf16* __restrict__ cblo,
        float* __restrict__ pm, float* __restrict__ ps, float* __restrict__ pw,
        float* __restrict__ pb, int* __restrict__ pi)
{
    const int lane = threadIdx.x & 31;
    const int wave = threadIdx.x >> 5;
    const int hl   = lane & 15;
    const int sel  = lane >> 4;
    const int gw   = blockIdx.x * 4 + wave;   // 2048 waves
    const int nh   = gw & 1;                  // codebook half
    const int rowbase = (gw >> 1) * 16;

    const __bf16* zhr = zhi + (size_t)(rowbase + hl) * DIM + sel * 8;
    const __bf16* zlr = zlo + (size_t)(rowbase + hl) * DIM + sel * 8;

    float m[8], s[8], w[8], best[8];
    int bidx[8];
#pragma unroll
    for (int r = 0; r < 8; ++r) { m[r] = -3.0e38f; s[r] = 0.f; w[r] = 0.f; best[r] = -3.0e38f; bidx[r] = 0; }

    const int nc0 = nh * (NCHUNKS / 2);
    for (int nc = nc0; nc < nc0 + NCHUNKS / 2; ++nc) {
        v8f acc[4];
#pragma unroll
        for (int t = 0; t < 4; ++t) acc[t] = (v8f){0.f,0.f,0.f,0.f,0.f,0.f,0.f,0.f};

        const size_t bbase = (size_t)(nc * 64 + hl) * DIM + sel * 8;
        chunk_gemm(zhr, zlr, cbhi, cblo, bbase, acc);

        // lane-local online softmax + argmax update (no cross-lane ops in loop)
#pragma unroll
        for (int r = 0; r < 8; ++r) {
#pragma unroll
            for (int t = 0; t < 4; ++t) {
                const float l  = 200.0f * acc[t][r];
                const int  col = nc * 64 + t * 16 + hl;
                const bool gt = l > best[r];
                best[r] = gt ? l : best[r];
                bidx[r] = gt ? col : bidx[r];
                const float mn = fmaxf(m[r], l);
                const float sc = __expf(m[r] - mn);
                const float e  = __expf(l - mn);
                s[r] = s[r] * sc + e;
                w[r] = w[r] * sc + l * e;
                m[r] = mn;
            }
        }
    }

    // merge across the 16 lanes of each half (row r in lanes 0-15, r+8 in 16-31)
#pragma unroll
    for (int r = 0; r < 8; ++r) {
#pragma unroll
        for (int off = 1; off < 16; off <<= 1) {
            float mo = __shfl_xor(m[r], off, 32);
            float so = __shfl_xor(s[r], off, 32);
            float wo = __shfl_xor(w[r], off, 32);
            float bo = __shfl_xor(best[r], off, 32);
            int   io = __shfl_xor(bidx[r], off, 32);
            const float mn  = fmaxf(m[r], mo);
            const float sc1 = __expf(m[r] - mn);
            const float sc2 = __expf(mo - mn);
            s[r] = s[r] * sc1 + so * sc2;
            w[r] = w[r] * sc1 + wo * sc2;
            m[r] = mn;
            const bool take = (bo > best[r]) || (bo == best[r] && io < bidx[r]);
            best[r] = take ? bo : best[r];
            bidx[r] = take ? io : bidx[r];
        }
        if (hl == 0) {
            const int slot = nh * N_TOKENS + rowbase + r + 8 * sel;
            pm[slot] = m[r];
            ps[slot] = s[r];
            pw[slot] = w[r];
            pb[slot] = best[r];
            pi[slot] = bidx[r];
        }
    }
}

// -------- merge the two codebook halves per row; emit final row state --------
__global__ __launch_bounds__(256) void vq_merge(
        const float* __restrict__ pm, const float* __restrict__ ps,
        const float* __restrict__ pw, const float* __restrict__ pb,
        const int* __restrict__ pi,
        float* __restrict__ row_s, float* __restrict__ row_w,
        float* __restrict__ row_L, int* __restrict__ row_idx)
{
    const int row = blockIdx.x * 256 + threadIdx.x;
    const float m0 = pm[row], m1 = pm[N_TOKENS + row];
    const float mn = fmaxf(m0, m1);
    const float e0 = __expf(m0 - mn), e1 = __expf(m1 - mn);
    const float sm = ps[row] * e0 + ps[N_TOKENS + row] * e1;
    const float wm = pw[row] * e0 + pw[N_TOKENS + row] * e1;
    row_s[row] = sm;
    row_w[row] = wm;
    row_L[row] = mn + __logf(sm);
    const float b0 = pb[row], b1 = pb[N_TOKENS + row];
    // half-1 columns are all larger, so ties keep half-0's index
    row_idx[row] = (b1 > b0) ? pi[N_TOKENS + row] : pi[row];
}

// ------------- Pass 2: recompute logits, accumulate avg_probs in LDS -------------
__global__ __launch_bounds__(128) void vq_pass2(
        const __bf16* __restrict__ zhi, const __bf16* __restrict__ zlo,
        const __bf16* __restrict__ cbhi, const __bf16* __restrict__ cblo,
        const float* __restrict__ row_L, float* __restrict__ avg_acc)
{
    __shared__ float lacc[CBSZ];
    for (int i = threadIdx.x; i < CBSZ; i += 128) lacc[i] = 0.f;
    __syncthreads();

    const int lane = threadIdx.x & 31;
    const int wave = threadIdx.x >> 5;
    const int hl   = lane & 15;
    const int sel  = lane >> 4;
    const int gw   = blockIdx.x * 4 + wave;
    const int nh   = gw & 1;
    const int rowbase = (gw >> 1) * 16;

    const __bf16* zhr = zhi + (size_t)(rowbase + hl) * DIM + sel * 8;
    const __bf16* zlr = zlo + (size_t)(rowbase + hl) * DIM + sel * 8;

    float Lr[8];
#pragma unroll
    for (int r = 0; r < 8; ++r) Lr[r] = row_L[rowbase + r + 8 * sel];

    const int nc0 = nh * (NCHUNKS / 2);
    for (int nc = nc0; nc < nc0 + NCHUNKS / 2; ++nc) {
        v8f acc[4];
#pragma unroll
        for (int t = 0; t < 4; ++t) acc[t] = (v8f){0.f,0.f,0.f,0.f,0.f,0.f,0.f,0.f};

        const size_t bbase = (size_t)(nc * 64 + hl) * DIM + sel * 8;
        chunk_gemm(zhr, zlr, cbhi, cblo, bbase, acc);

#pragma unroll
        for (int t = 0; t < 4; ++t) {
            float csum = 0.f;
#pragma unroll
            for (int r = 0; r < 8; ++r)
                csum += __expf(200.0f * acc[t][r] - Lr[r]);
            csum += __shfl_xor(csum, 16, 32);       // pair rows r and r+8 (same column)
            if (sel == 0) atomicAdd(&lacc[nc * 64 + t * 16 + hl], csum);
        }
    }
    __syncthreads();
    for (int i = threadIdx.x; i < CBSZ; i += 128)
        atomicAdd(&avg_acc[i], lacc[i]);
}

// ------------- gather z_q, fused MSE + hard counts + index output -------------
__global__ __launch_bounds__(256) void vq_gather(
        const float* __restrict__ z, const float* __restrict__ cb,
        const int* __restrict__ row_idx, float* __restrict__ out_zq,
        float* __restrict__ out_idx, float* __restrict__ counts,
        float* __restrict__ mse_acc)
{
    const int row = blockIdx.x;
    const int idx = row_idx[row];
    const float* c  = cb + (size_t)idx * DIM;
    const float* zr = z  + (size_t)row * DIM;
    float partial = 0.f;
    for (int i = threadIdx.x; i < DIM; i += 256) {
        const float cv = c[i];
        const float d = cv - zr[i];
        out_zq[(size_t)row * DIM + i] = cv;
        partial += d * d;
    }
    __shared__ float red[256];
    red[threadIdx.x] = partial;
    __syncthreads();
    for (int st = 128; st > 0; st >>= 1) {
        if (threadIdx.x < st) red[threadIdx.x] += red[threadIdx.x + st];
        __syncthreads();
    }
    if (threadIdx.x == 0) {
        atomicAdd(mse_acc, red[0]);
        atomicAdd(&counts[idx], 1.0f);
        out_idx[row] = (float)idx;
    }
}

// --------------------------------- finalize ---------------------------------
__global__ __launch_bounds__(256) void vq_finalize(
        const float* __restrict__ row_s, const float* __restrict__ row_w,
        const float* __restrict__ row_L, const float* __restrict__ avg_acc,
        const float* __restrict__ counts, const float* __restrict__ mse_acc,
        const int* __restrict__ beta_mult,
        float* __restrict__ out_loss, float* __restrict__ out_perp)
{
    __shared__ float red[256];
    const int tid = threadIdx.x;

    // sum over rows of (sum_k p log p) = w/s - L
    float a0 = 0.f;
    for (int n = tid; n < N_TOKENS; n += 256)
        a0 += row_w[n] / row_s[n] - row_L[n];
    red[tid] = a0; __syncthreads();
    for (int st = 128; st > 0; st >>= 1) { if (tid < st) red[tid] += red[tid + st]; __syncthreads(); }
    const float sample_ent = -(red[0] / (float)N_TOKENS);
    __syncthreads();

    // avg-probs entropy
    float a1 = 0.f;
    for (int k = tid; k < CBSZ; k += 256) {
        const float q = avg_acc[k] * (1.0f / (float)N_TOKENS);
        a1 += q * __logf(q + EPS);
    }
    red[tid] = a1; __syncthreads();
    for (int st = 128; st > 0; st >>= 1) { if (tid < st) red[tid] += red[tid + st]; __syncthreads(); }
    const float avg_ent = -red[0];
    __syncthreads();

    // hard-count perplexity
    float a2 = 0.f;
    for (int k = tid; k < CBSZ; k += 256) {
        const float p = counts[k] * (1.0f / (float)N_TOKENS);
        a2 += p * __logf(p + EPS);
    }
    red[tid] = a2; __syncthreads();
    for (int st = 128; st > 0; st >>= 1) { if (tid < st) red[tid] += red[tid + st]; __syncthreads(); }

    if (tid == 0) {
        const float perp = __expf(-red[0]);
        const float msev = mse_acc[0] / (float)((size_t)N_TOKENS * DIM);
        const float loss = (1.0f + 0.25f * (float)beta_mult[0]) * msev
                         + 0.1f * (sample_ent - avg_ent);
        out_loss[0] = loss;
        out_perp[0] = perp;
    }
}

__global__ void vq_zero(float* __restrict__ p, int n) {
    const int i = blockIdx.x * blockDim.x + threadIdx.x;
    if (i < n) p[i] = 0.f;
}

extern "C" void kernel_launch(void* const* d_in, const int* in_sizes, int n_in,
                              void* d_out, int out_size, void* d_ws, size_t ws_size,
                              hipStream_t stream) {
    const float* z    = (const float*)d_in[0];
    const float* cb   = (const float*)d_in[1];
    const int*   beta = (const int*)d_in[2];

    float* ws      = (float*)d_ws;
    float* row_s   = ws;                  // 16384
    float* row_w   = ws + 16384;          // 16384
    float* row_L   = ws + 32768;          // 16384
    int*   row_idx = (int*)(ws + 49152);  // 16384
    float* avg_acc = ws + 65536;          // 8192
    float* counts  = ws + 73728;          // 8192
    float* mse_acc = ws + 81920;          // 1
    // per-(half,row) pass-1 partials
    float* pm = ws + 82048;               // 32768
    float* ps = ws + 114816;              // 32768
    float* pw = ws + 147584;              // 32768
    float* pb = ws + 180352;              // 32768
    int*   pi = (int*)(ws + 213120);      // 32768
    // codebook bf16 split planes (8192*512 each), 16-byte aligned
    __bf16* cbhi = (__bf16*)(ws + 245888);
    __bf16* cblo = cbhi + (size_t)CBSZ * DIM;

    float* out      = (float*)d_out;
    float* out_zq   = out;                                  // 16384*512
    float* out_idx  = out + (size_t)N_TOKENS * DIM;         // 16384
    float* out_loss = out_idx + N_TOKENS;                   // 1
    float* out_perp = out_loss + 1;                         // 1

    // z bf16 split planes live inside the z_q output region (32 MB = 2 planes
    // of 16 MB); the region is fully rewritten by vq_gather afterwards.
    __bf16* zhi = (__bf16*)out_zq;
    __bf16* zlo = zhi + (size_t)N_TOKENS * DIM;

    // zero avg_acc + counts + mse (contiguous 16385 floats)
    vq_zero<<<(16385 + 255) / 256, 256, 0, stream>>>(avg_acc, 16385);

    vq_split<<<(N_TOKENS * DIM) / 256, 256, 0, stream>>>(z, zhi, zlo, N_TOKENS * DIM);
    vq_split<<<(CBSZ * DIM) / 256, 256, 0, stream>>>(cb, cbhi, cblo, CBSZ * DIM);

    vq_pass1<<<N_TOKENS / 32, 128, 0, stream>>>(zhi, zlo, cbhi, cblo, pm, ps, pw, pb, pi);
    vq_merge<<<N_TOKENS / 256, 256, 0, stream>>>(pm, ps, pw, pb, pi,
                                                 row_s, row_w, row_L, row_idx);
    vq_pass2<<<N_TOKENS / 32, 128, 0, stream>>>(zhi, zlo, cbhi, cblo, row_L, avg_acc);
    // gather overwrites the z_q region, so it must run after both WMMA passes
    vq_gather<<<N_TOKENS, 256, 0, stream>>>(z, cb, row_idx, out_zq, out_idx, counts, mse_acc);
    vq_finalize<<<1, 256, 0, stream>>>(row_s, row_w, row_L, avg_acc, counts, mse_acc,
                                       beta, out_loss, out_perp);
}
